// GNNMultiheadAttnDrugPooling_1675037245811
// MI455X (gfx1250) — compile-verified, matrix operand order checked by reading, not run on hardware
//
#include <hip/hip_runtime.h>
#include <hip/hip_bf16.h>

#define E_DIM 256
#define HD_DIM 256
#define O_DIM 256
#define H_DIM 4
// swizzled fragment store: [h][ct(16)][ks(8)][lane(32)][e(16)] bf16
#define PER_MAT (H_DIM * 16 * 8 * 32 * 16)   // 262144 elements per weight matrix

typedef __attribute__((ext_vector_type(16))) __bf16 v16bf;
typedef __attribute__((ext_vector_type(8)))  float  v8f;

// K index inside a 32-wide K-step for 16-bit A/B fragments (ISA 7.12.2)
__device__ __forceinline__ int frag_k(int e, int kb) {
  return ((e < 8) ? e : e + 8) + kb;
}

// A fragment from row-major f32 source (one row per lane, cvt to bf16)
__device__ __forceinline__ v16bf load_a_rowmajor(const float* __restrict__ base, int kb) {
  v16bf a;
#pragma unroll
  for (int e = 0; e < 16; e += 2) {
    const int k = frag_k(e, kb);
    const float2 f = *(const float2*)(base + k);
    a[e]     = (__bf16)f.x;
    a[e + 1] = (__bf16)f.y;
  }
  return a;
}

// B fragment from pre-swizzled bf16 weights: one contiguous 32B chunk per lane
__device__ __forceinline__ v16bf load_frag_sw(const __bf16* __restrict__ mat_h,
                                              int ct, int ks, int lane) {
  return *(const v16bf*)(mat_h + (size_t)(((ct * 8 + ks) * 32 + lane) * 16));
}

// ---------------------------------------------------------------------------
// Prep: convert W1g, W1v, W2v (f32 [H,256,256]) into fragment-major bf16.
// One thread per output element; output index encodes (mat,h,ct,ks,lane,e).
// ---------------------------------------------------------------------------
__global__ void __launch_bounds__(256) swizzle_weights_kernel(
    const float* __restrict__ W1g, const float* __restrict__ W1v,
    const float* __restrict__ W2v, __bf16* __restrict__ out) {
  const int idx = blockIdx.x * blockDim.x + threadIdx.x;
  if (idx >= 3 * PER_MAT) return;
  const int mat = idx / PER_MAT;
  const int rem = idx % PER_MAT;
  const int h   = rem >> 16;           // / 65536
  const int off = rem & 65535;
  const int e    = off & 15;
  const int lane = (off >> 4) & 31;
  const int ks   = (off >> 9) & 7;
  const int ct   = off >> 12;
  const int col  = ct * 16 + (lane & 15);
  const int k    = frag_k(e, (lane >> 4) * 8) + ks * 32;
  const float* src = (mat == 0) ? W1g : (mat == 1) ? W1v : W2v;
  out[idx] = (__bf16)src[((size_t)h * 256 + k) * 256 + col];
}

// ---------------------------------------------------------------------------
// Kernel 1: gate[h,n] = relu(x@W1g[h] + b1g[h]) . W2g[h] + b2g[h]
// ---------------------------------------------------------------------------
__global__ void __launch_bounds__(256) gate_kernel(
    const float* __restrict__ x, const __bf16* __restrict__ W1g_b,
    const float* __restrict__ b1g, const float* __restrict__ W2g,
    const float* __restrict__ b2g, float* __restrict__ gate, int N) {
  const int lane = threadIdx.x & 31;
  const int wave = threadIdx.x >> 5;
  const int tile = blockIdx.x * 8 + wave;
  const int row0 = tile * 16;
  if (row0 >= N) return;
  const int rl  = lane & 15;
  const int hiL = lane >> 4;
  const int kb  = hiL * 8;

  const int rowA = min(row0 + rl, N - 1);
  const float* xrow = x + (size_t)rowA * E_DIM;
  v16bf afrag[8];
#pragma unroll
  for (int ks = 0; ks < 8; ++ks) afrag[ks] = load_a_rowmajor(xrow + ks * 32, kb);

  for (int h = 0; h < H_DIM; ++h) {
    float gpart[8];
#pragma unroll
    for (int r = 0; r < 8; ++r) gpart[r] = 0.f;
    const __bf16* w1 = W1g_b + (size_t)h * (16 * 8 * 32 * 16);
    for (int ct = 0; ct < HD_DIM / 16; ++ct) {
      const int col = ct * 16 + rl;
      v8f c = {};
#pragma unroll
      for (int ks = 0; ks < 8; ++ks) {
        v16bf bfrag = load_frag_sw(w1, ct, ks, lane);
        c = __builtin_amdgcn_wmma_f32_16x16x32_bf16(false, afrag[ks], false, bfrag,
                                                    (short)0, c, false, false);
      }
      const float b1 = b1g[h * HD_DIM + col];
      const float w2 = W2g[h * HD_DIM + col];
#pragma unroll
      for (int r = 0; r < 8; ++r) {
        float v = c[r] + b1;
        v = v > 0.f ? v : 0.f;
        gpart[r] += v * w2;   // fold hg . W2g immediately (per-lane partial)
      }
    }
#pragma unroll
    for (int r = 0; r < 8; ++r) {
      float s = gpart[r];
      s += __shfl_xor(s, 1, 32);
      s += __shfl_xor(s, 2, 32);
      s += __shfl_xor(s, 4, 32);
      s += __shfl_xor(s, 8, 32);
      gpart[r] = s;
    }
    if (rl == 0) {
      const float bb = b2g[h];
#pragma unroll
      for (int r = 0; r < 8; ++r) {
        const int rr = row0 + r + hiL * 8;
        if (rr < N) gate[(size_t)h * N + rr] = gpart[r] + bb;
      }
    }
  }
}

// ---------------------------------------------------------------------------
// Kernel 2: per (head, graph) segment max + sum(exp) via binary search
// ---------------------------------------------------------------------------
__global__ void stats_kernel(const float* __restrict__ gate, const int* __restrict__ batch,
                             float* __restrict__ mOut, float* __restrict__ dOut,
                             int N, int G) {
  const int g = blockIdx.x % G;
  const int h = blockIdx.x / G;
  const int lane = threadIdx.x;
  int lo = 0, hi2 = N;
  while (lo < hi2) { const int mid = (lo + hi2) >> 1; if (batch[mid] < g) lo = mid + 1; else hi2 = mid; }
  const int start = lo;
  hi2 = N;
  while (lo < hi2) { const int mid = (lo + hi2) >> 1; if (batch[mid] <= g) lo = mid + 1; else hi2 = mid; }
  const int end = lo;

  const float* gp = gate + (size_t)h * N;
  float mx = -3.402823466e38f;
  for (int i = start + lane; i < end; i += 32) mx = fmaxf(mx, gp[i]);
#pragma unroll
  for (int off = 16; off > 0; off >>= 1) mx = fmaxf(mx, __shfl_xor(mx, off, 32));
  float s = 0.f;
  for (int i = start + lane; i < end; i += 32) s += __expf(gp[i] - mx);
#pragma unroll
  for (int off = 16; off > 0; off >>= 1) s += __shfl_xor(s, off, 32);
  if (lane == 0) { mOut[h * G + g] = mx; dOut[h * G + g] = s; }
}

// ---------------------------------------------------------------------------
// Kernel 3: one block (8 waves) per graph.
//  phase 1: hv = relu(x@W1v+b1v) -> LDS in A-fragment-major bf16 layout
//  phase 2: val = hv@W2v (A frags = contiguous b128 LDS loads), softmax-
//           weighted row-sum into register accumulators.
// ---------------------------------------------------------------------------
__global__ void __launch_bounds__(256) pool_kernel(
    const float* __restrict__ x, const int* __restrict__ batch,
    const __bf16* __restrict__ W1v_b, const float* __restrict__ b1v,
    const __bf16* __restrict__ W2v_b, const float* __restrict__ b2v,
    const float* __restrict__ gate, const float* __restrict__ mIn,
    const float* __restrict__ dIn, float* __restrict__ out, int N, int G) {
  __shared__ __bf16 hv_frag[H_DIM * 8 * 32 * 16];  // [h][ks][lane][e], 32 KB
  __shared__ float  pooled_s[H_DIM][O_DIM];        // 4 KB
  __shared__ float  arow_s[H_DIM][16];

  const int g = blockIdx.x;
  const int tid = threadIdx.x;

  int lo = 0, hi2 = N;
  while (lo < hi2) { const int mid = (lo + hi2) >> 1; if (batch[mid] < g) lo = mid + 1; else hi2 = mid; }
  const int start = lo;
  hi2 = N;
  while (lo < hi2) { const int mid = (lo + hi2) >> 1; if (batch[mid] <= g) lo = mid + 1; else hi2 = mid; }
  const int end = lo;

  if (start >= end) {   // empty graph: segment_sum over nothing == 0
    for (int c2 = tid; c2 < O_DIM; c2 += blockDim.x) out[(size_t)g * O_DIM + c2] = 0.f;
    return;
  }

  const int lane = tid & 31;
  const int wave = tid >> 5;
  const int rl  = lane & 15;
  const int hiL = lane >> 4;
  const int kb  = hiL * 8;

  float acc[8];
#pragma unroll
  for (int j = 0; j < 8; ++j) acc[j] = 0.f;

  const int ntiles = (end - start + 15) / 16;
  for (int t = 0; t < ntiles; ++t) {
    const int row0 = start + t * 16;
    if (tid < H_DIM * 16) {  // softmax weights for this tile's rows, per head
      const int h = tid >> 4;
      const int ridx = tid & 15;
      const int node = row0 + ridx;
      float a = 0.f;
      if (node < end)
        a = __expf(gate[(size_t)h * N + node] - mIn[h * G + g]) / dIn[h * G + g];
      arow_s[h][ridx] = a;
    }
    __syncthreads();

    // phase 1: hv -> LDS (fragment-major scatter)
    const int rowA = min(row0 + rl, N - 1);
    const float* xrow = x + (size_t)rowA * E_DIM;
    v16bf afrag[8];
#pragma unroll
    for (int ks = 0; ks < 8; ++ks) afrag[ks] = load_a_rowmajor(xrow + ks * 32, kb);
#pragma unroll
    for (int j = 0; j < 8; ++j) {
      const int task = wave * 8 + j;       // 64 tasks = (h, col-tile)
      const int h  = task >> 4;
      const int ct = task & 15;
      const int col = ct * 16 + rl;
      const __bf16* w1 = W1v_b + (size_t)h * (16 * 8 * 32 * 16);
      v8f c = {};
#pragma unroll
      for (int ks = 0; ks < 8; ++ks) {
        v16bf bfrag = load_frag_sw(w1, ct, ks, lane);
        c = __builtin_amdgcn_wmma_f32_16x16x32_bf16(false, afrag[ks], false, bfrag,
                                                    (short)0, c, false, false);
      }
      const float b1 = b1v[h * HD_DIM + col];
      // scatter into the A-fragment layout for phase 2:
      // k = col; ks2 = k>>5; hi_c = bit3(k); e = (k&7) + 8*((k%32)>=16)
      const int ks2 = ct >> 1;
      const int ee  = (rl & 7) + ((ct & 1) << 3);
      const int hic = (rl >> 3) & 1;
#pragma unroll
      for (int r = 0; r < 8; ++r) {
        float v = c[r] + b1;
        v = v > 0.f ? v : 0.f;
        const int lane_c = (r + hiL * 8) + (hic << 4);
        hv_frag[(((h * 8 + ks2) * 32) + lane_c) * 16 + ee] = (__bf16)v;
      }
    }
    __syncthreads();

    // phase 2: val = hv @ W2v, weighted row-sum
#pragma unroll
    for (int j = 0; j < 8; ++j) {
      const int task = wave * 8 + j;       // 64 tasks = (h, O col-tile)
      const int h   = task >> 4;
      const int oct = task & 15;
      const __bf16* w2 = W2v_b + (size_t)h * (16 * 8 * 32 * 16);
      v8f c = {};
#pragma unroll
      for (int ks = 0; ks < 8; ++ks) {
        v16bf a2 = *(const v16bf*)&hv_frag[((h * 8 + ks) * 32 + lane) * 16];
        v16bf bfrag = load_frag_sw(w2, oct, ks, lane);
        c = __builtin_amdgcn_wmma_f32_16x16x32_bf16(false, a2, false, bfrag,
                                                    (short)0, c, false, false);
      }
      float s = 0.f;
#pragma unroll
      for (int r = 0; r < 8; ++r) s += arow_s[h][r + hiL * 8] * c[r];
      s += __shfl_xor(s, 16, 32);          // combine the two M halves
      acc[j] += s;
    }
    __syncthreads();
  }

  // unique (h, oct) ownership per wave-task -> plain stores
#pragma unroll
  for (int j = 0; j < 8; ++j) {
    const int task = wave * 8 + j;
    const int h   = task >> 4;
    const int oct = task & 15;
    if (hiL == 0) pooled_s[h][oct * 16 + rl] = acc[j];
  }
  __syncthreads();
  for (int c2 = tid; c2 < O_DIM; c2 += blockDim.x) {
    float s = 0.f;
#pragma unroll
    for (int h = 0; h < H_DIM; ++h) s += pooled_s[h][c2] + b2v[h * O_DIM + c2];
    out[(size_t)g * O_DIM + c2] = s * (1.0f / H_DIM);
  }
}

extern "C" void kernel_launch(void* const* d_in, const int* in_sizes, int n_in,
                              void* d_out, int out_size, void* d_ws, size_t ws_size,
                              hipStream_t stream) {
  const float* x   = (const float*)d_in[0];
  const int*   bat = (const int*)  d_in[1];
  const float* W1g = (const float*)d_in[2];
  const float* b1g = (const float*)d_in[3];
  const float* W2g = (const float*)d_in[4];
  const float* b2g = (const float*)d_in[5];
  const float* W1v = (const float*)d_in[6];
  const float* b1v = (const float*)d_in[7];
  const float* W2v = (const float*)d_in[8];
  const float* b2v = (const float*)d_in[9];

  const int N = in_sizes[1];          // batch has N elements
  const int G = out_size / O_DIM;

  // workspace layout (64B-aligned slices)
  char* ws = (char*)d_ws;
  size_t off = 0;
  float* gate = (float*)(ws + off); off += (size_t)H_DIM * N * sizeof(float); off = (off + 63) & ~(size_t)63;
  float* m    = (float*)(ws + off); off += (size_t)H_DIM * G * sizeof(float); off = (off + 63) & ~(size_t)63;
  float* dden = (float*)(ws + off); off += (size_t)H_DIM * G * sizeof(float); off = (off + 63) & ~(size_t)63;
  __bf16* wsw = (__bf16*)(ws + off);              // 3 * PER_MAT bf16 = 1.5 MB
  __bf16* W1g_b = wsw;
  __bf16* W1v_b = wsw + PER_MAT;
  __bf16* W2v_b = wsw + 2 * (size_t)PER_MAT;

  swizzle_weights_kernel<<<(3 * PER_MAT + 255) / 256, 256, 0, stream>>>(W1g, W1v, W2v, wsw);
  const int tiles = (N + 15) / 16;
  gate_kernel<<<(tiles + 7) / 8, 256, 0, stream>>>(x, W1g_b, b1g, W2g, b2g, gate, N);
  stats_kernel<<<H_DIM * G, 32, 0, stream>>>(gate, bat, m, dden, N, G);
  pool_kernel<<<G, 256, 0, stream>>>(x, bat, W1v_b, b1v, W2v_b, b2v, gate, m, dden,
                                     (float*)d_out, N, G);
}
